// AtomicDeformationNNConv_63127429317290
// MI455X (gfx1250) — matrix operand order repaired and checked by reading, not compiled, not run on hardware
//
#include <hip/hip_runtime.h>
#include <stdint.h>

typedef __attribute__((ext_vector_type(16))) _Float16 v16h;
typedef __attribute__((ext_vector_type(8)))  float    v8f;

#define TPB 256
#define MEDGE 32   // edges per workgroup (two 16-row WMMA A blocks)

// ---------------------------------------------------------------- utilities
__global__ void zero_f32_kernel(float* p, int n) {
  int i = blockIdx.x * TPB + threadIdx.x;
  if (i < n) p[i] = 0.0f;
}
__global__ void zero_i32_kernel(int* p, int n) {
  int i = blockIdx.x * TPB + threadIdx.x;
  if (i < n) p[i] = 0;
}
__global__ void count_edges_kernel(const int* __restrict__ dst, int E, int* __restrict__ cnt) {
  int i = blockIdx.x * TPB + threadIdx.x;
  if (i < E) atomicAdd(&cnt[dst[i]], 1);
}

// ------------------------------------------------------------ W2 pre-swizzle
// Pack w2 (fp32 [Kh,F]) + b2 (appended as K-row "Kh", rows Kh+1..Kh+31 zero)
// into f16 per-lane B fragments for V_WMMA_F32_16X16X32_F16 (contiguous k, same order as A):
// element e of lane L for (col-tile t, k-chunk c):  B[k = c*32 + (L>>4)*16 + e][f = t*16 + (L&15)]
// stored at swz[((t*C + c)*32 + L)*16 + e]  -> inner loop does 2 coalesced b128 loads/lane.
__global__ void prep_swz_kernel(const float* __restrict__ w2, const float* __restrict__ b2,
                                int Kh, int F, int C, _Float16* __restrict__ swz) {
  int idx = blockIdx.x * TPB + threadIdx.x;
  int total = (F >> 4) * C * 512;
  if (idx >= total) return;
  int t  = idx / (C * 512);
  int r  = idx - t * (C * 512);
  int c  = r >> 9;
  int r2 = r & 511;
  int L  = r2 >> 4;
  int e  = r2 & 15;
  int hi = L >> 4;
  int n  = L & 15;
  int f  = t * 16 + n;
  int kk = c * 32 + hi * 16 + e;
  float v = 0.0f;
  if (kk < Kh)       v = w2[(long long)kk * F + f];
  else if (kk == Kh) v = b2[f];
  swz[idx] = (_Float16)v;
}

// ------------------------------------------------------- fused NNConv GEMM
// One workgroup = 32 edges (two 16-row A blocks resident in LDS, ~264 KB for
// the hidden layers -- CDNA5's 320 KB WGP LDS makes this legal). 8 waves sweep
// the F/16 column tiles of theta = [relu(a*w1+b1), 1] @ [w2; b2]; each B
// fragment fetched from L2 feeds TWO wmmas (32 FLOP per L2 byte). The fused
// epilogue contracts each theta tile with x[src] (LDS) into per-edge messages,
// then scatters to per-node sums.
__global__ void __launch_bounds__(TPB)
nnconv_fused_kernel(const float* __restrict__ hin, const float* __restrict__ attr,
                    const int* __restrict__ src, const int* __restrict__ dst,
                    const float* __restrict__ w1, const float* __restrict__ b1,
                    const _Float16* __restrict__ swz, float* __restrict__ nodeSum,
                    int E, int I, int O, int Kh, int C, int T)
{
  extern __shared__ _Float16 Ah[];        // 2 * C * 512 halves: A fragments (32 x (Kh+32))
  __shared__ float xtile[MEDGE * 64];     // gathered x[src] rows (I <= 64)
  __shared__ float msg[MEDGE * 64];       // per-edge message accumulators (O <= 64)
  __shared__ float attrS[MEDGE];

  const int tid = threadIdx.x;
  const int e0  = blockIdx.x * MEDGE;

  if (tid < MEDGE) {
    int ei = e0 + tid; if (ei > E - 1) ei = E - 1;
    attrS[tid] = attr[ei];
  }
  __syncthreads();

  // build A fragments in LDS in per-lane fragment order (contiguous k, same as B):
  // entry (blk*C*512 + c*512 + L*16 + e) =
  //   h1[row m = blk*16 + (L&15)][k = c*32 + (L>>4)*16 + e]
  const int tot = 2 * C * 512;
  for (int idx = tid; idx < tot; idx += TPB) {
    int blk = idx / (C * 512);
    int r   = idx - blk * (C * 512);
    int c   = r >> 9;
    int r2  = r & 511;
    int L   = r2 >> 4;
    int e   = r2 & 15;
    int hi  = L >> 4;
    int m   = blk * 16 + (L & 15);
    int kk  = c * 32 + hi * 16 + e;
    float v;
    if (kk < Kh) {
      v = fmaf(attrS[m], w1[kk], b1[kk]);
      v = fmaxf(v, 0.0f);
    } else {
      v = (kk == Kh) ? 1.0f : 0.0f;   // bias row of the augmented GEMM
    }
    Ah[idx] = (_Float16)v;
  }
  // gather x[src] rows + clear message accumulators
  for (int p = tid; p < MEDGE * I; p += TPB) {
    int m = p / I, i = p - m * I;
    int ei = e0 + m; if (ei > E - 1) ei = E - 1;
    xtile[p] = hin[(long long)src[ei] * I + i];
  }
  for (int p = tid; p < MEDGE * O; p += TPB) msg[p] = 0.0f;
  __syncthreads();

  const int wv = tid >> 5, lane = tid & 31;
  const int n = lane & 15, hi = lane >> 4;
  const int blkStride = C * 64;           // uint4 stride between the two A blocks

  typedef union { v16h v; uint4 q[2]; } frag_t;

  for (int t = wv; t < T; t += 8) {
    v8f acc0 = {};
    v8f acc1 = {};
    const uint4* Bq = (const uint4*)(swz + (size_t)t * C * 512);
    const uint4* Aq = (const uint4*)Ah;
    for (int c = 0; c < C; ++c) {
      frag_t a0, a1, b;
      int aoff = (c * 32 + lane) * 2;
      a0.q[0] = Aq[aoff];             a0.q[1] = Aq[aoff + 1];             // ds_load_b128
      a1.q[0] = Aq[blkStride + aoff]; a1.q[1] = Aq[blkStride + aoff + 1];
      b.q[0]  = Bq[aoff];             b.q[1]  = Bq[aoff + 1];             // global_load_b128 (L2)
      if (c + 1 < C)
        __builtin_prefetch((const void*)(Bq + aoff + 64), 0, 3);          // next k-chunk
      acc0 = __builtin_amdgcn_wmma_f32_16x16x32_f16(
          false, a0.v, false, b.v, (short)0, acc0, false, false);
      acc1 = __builtin_amdgcn_wmma_f32_16x16x32_f16(
          false, a1.v, false, b.v, (short)0, acc1, false, false);
    }
    // epilogue: theta[m][f] -> msg[m][o] += x[m][i] * theta,  f = i*O + o
    int f = t * 16 + n;
    int i = f / O;
    int o = f - i * O;
#pragma unroll
    for (int r = 0; r < 8; ++r) {
      int m0 = r + hi * 8;                       // C/D layout: VGPR r, half hi
      int m1 = 16 + m0;
      atomicAdd(&msg[m0 * O + o], xtile[m0 * I + i] * acc0[r]);   // ds_add_f32
      atomicAdd(&msg[m1 * O + o], xtile[m1 * I + i] * acc1[r]);
    }
  }
  __syncthreads();

  for (int p = tid; p < MEDGE * O; p += TPB) {
    int m = p / O, o = p - m * O;
    int ei = e0 + m;
    if (ei < E)
      unsafeAtomicAdd(&nodeSum[(long long)dst[ei] * O + o], msg[p]); // global_atomic_add_f32
  }
}

// -------------------------------------------------- per-node epilogue
// out = nodeSum/max(cnt,1) + hin@root + bias  [+ BN -> relu] [+ residual]
__global__ void node_epilogue_kernel(const float* __restrict__ nodeSum, const int* __restrict__ cnt,
                                     const float* __restrict__ hin, const float* __restrict__ root,
                                     const float* __restrict__ bias,
                                     const float* __restrict__ bng, const float* __restrict__ bnb,
                                     const float* __restrict__ bnm, const float* __restrict__ bnv,
                                     const float* __restrict__ res, float* __restrict__ out,
                                     int N, int I, int O, int useBn)
{
  int idx = blockIdx.x * TPB + threadIdx.x;
  if (idx >= N * O) return;
  int nn = idx / O, o = idx - nn * O;
  int c = cnt[nn]; if (c < 1) c = 1;
  float v = nodeSum[idx] / (float)c;
  float d = 0.0f;
  for (int i = 0; i < I; ++i) d = fmaf(hin[(long long)nn * I + i], root[i * O + o], d);
  v += d + bias[o];
  if (useBn) {
    v = (v - bnm[o]) * rsqrtf(bnv[o] + 1e-5f) * bng[o] + bnb[o];
    v = fmaxf(v, 0.0f);
  }
  if (res) v += res[idx];
  out[idx] = v;
}

// ---------------------------------------------------------------- driver
extern "C" void kernel_launch(void* const* d_in, const int* in_sizes, int n_in,
                              void* d_out, int out_size, void* d_ws, size_t ws_size,
                              hipStream_t stream) {
  const float* x       = (const float*)d_in[0];
  const int*   eidx    = (const int*)  d_in[1];
  const float* eattr   = (const float*)d_in[2];
  const float* in_w1   = (const float*)d_in[3];
  const float* in_b1   = (const float*)d_in[4];
  const float* in_w2   = (const float*)d_in[5];
  const float* in_b2   = (const float*)d_in[6];
  const float* in_root = (const float*)d_in[7];
  const float* in_bias = (const float*)d_in[8];
  const float* in_bn_g = (const float*)d_in[9];
  const float* in_bn_b = (const float*)d_in[10];
  const float* in_bn_m = (const float*)d_in[11];
  const float* in_bn_v = (const float*)d_in[12];
  const float* h_w1    = (const float*)d_in[13];
  const float* h_b1    = (const float*)d_in[14];
  const float* h_w2    = (const float*)d_in[15];
  const float* h_b2    = (const float*)d_in[16];
  const float* h_root  = (const float*)d_in[17];
  const float* h_bias  = (const float*)d_in[18];
  const float* h_bn_g  = (const float*)d_in[19];
  const float* h_bn_b  = (const float*)d_in[20];
  const float* h_bn_m  = (const float*)d_in[21];
  const float* h_bn_v  = (const float*)d_in[22];
  const float* o_w1    = (const float*)d_in[23];
  const float* o_b1    = (const float*)d_in[24];
  const float* o_w2    = (const float*)d_in[25];
  const float* o_b2    = (const float*)d_in[26];
  const float* o_root  = (const float*)d_in[27];
  const float* o_bias  = (const float*)d_in[28];

  const int N = in_sizes[0] / 10;
  const int E = in_sizes[1] / 2;
  const int* srcP = eidx;
  const int* dstP = eidx + E;

  // workspace carve-up (max swz = hidden layer: 256 tiles * 129 chunks * 512 halves)
  char* ws = (char*)d_ws;
  _Float16* swz = (_Float16*)ws;
  size_t off = (size_t)256 * 129 * 512 * 2;
  int* cnt = (int*)(ws + off);         off += (size_t)N * 4;      off = (off + 255) & ~(size_t)255;
  float* nodeSum = (float*)(ws + off); off += (size_t)N * 64 * 4; off = (off + 255) & ~(size_t)255;
  float* hA = (float*)(ws + off);      off += (size_t)N * 64 * 4; off = (off + 255) & ~(size_t)255;
  float* hB = (float*)(ws + off);

  zero_i32_kernel<<<(N + TPB - 1) / TPB, TPB, 0, stream>>>(cnt, N);
  count_edges_kernel<<<(E + TPB - 1) / TPB, TPB, 0, stream>>>(dstP, E, cnt);

  auto layer = [&](const float* hin, int I, int O, int Kh,
                   const float* w1, const float* b1, const float* w2, const float* b2,
                   const float* root, const float* bias,
                   const float* g, const float* bb, const float* m, const float* v,
                   const float* res, float* outp) {
    const int F = I * O;                 // 640 / 4096 / 192: all multiples of 16
    const int C = (Kh + 32) / 32;        // K chunks incl. appended bias row block
    const int T = F >> 4;
    const int prepTot = T * C * 512;
    prep_swz_kernel<<<(prepTot + TPB - 1) / TPB, TPB, 0, stream>>>(w2, b2, Kh, F, C, swz);
    zero_f32_kernel<<<(N * O + TPB - 1) / TPB, TPB, 0, stream>>>(nodeSum, N * O);
    size_t lds = (size_t)2 * C * 512 * sizeof(_Float16);  // up to ~264 KB (CDNA5: 320 KB/WGP)
    nnconv_fused_kernel<<<(E + MEDGE - 1) / MEDGE, TPB, lds, stream>>>(
        hin, eattr, srcP, dstP, w1, b1, swz, nodeSum, E, I, O, Kh, C, T);
    node_epilogue_kernel<<<(N * O + TPB - 1) / TPB, TPB, 0, stream>>>(
        nodeSum, cnt, hin, root, bias, g, bb, m, v, res, outp, N, I, O, g != nullptr);
  };

  // input conv + BN + relu
  layer(x, 10, 64, 640, in_w1, in_b1, in_w2, in_b2, in_root, in_bias,
        in_bn_g, in_bn_b, in_bn_m, in_bn_v, nullptr, hA);
  // hidden residual block 0
  layer(hA, 64, 64, 4096, h_w1, h_b1, h_w2, h_b2, h_root, h_bias,
        h_bn_g, h_bn_b, h_bn_m, h_bn_v, hA, hB);
  // hidden residual block 1
  layer(hB, 64, 64, 4096,
        h_w1 + 4096, h_b1 + 4096, h_w2 + (size_t)4096 * 4096, h_b2 + 4096,
        h_root + 64 * 64, h_bias + 64,
        h_bn_g + 64, h_bn_b + 64, h_bn_m + 64, h_bn_v + 64, hB, hA);
  // output conv (no BN / relu / residual)
  layer(hA, 64, 3, 192, o_w1, o_b1, o_w2, o_b2, o_root, o_bias,
        nullptr, nullptr, nullptr, nullptr, nullptr, (float*)d_out);
}
